// PDselector_56100862820521
// MI455X (gfx1250) — compile-verified, matrix operand order checked by reading, not run on hardware
//
#include <hip/hip_runtime.h>
#include <hip/hip_bf16.h>
#include <math.h>

#define B_    8
#define S_    4096
#define D_    3
#define P_    400
#define KP_   416      /* K padded to 13 * 32 for wmma 16x16x32 */
#define TS_   32       /* samples per block in k_main */
#define INHIB_   0.1f
#define ENDRATE_ 0.05f

typedef __bf16 bf16;
typedef __attribute__((ext_vector_type(16))) __bf16 v16bf;
typedef __attribute__((ext_vector_type(8)))  __bf16 v8bf;
typedef __attribute__((ext_vector_type(8)))  float  v8f;
typedef __attribute__((ext_vector_type(4)))  float  v4f;

/* workspace layout (bytes) */
#define WS_WB_OFF    0                        /* bf16 [400][416] = 332800 B  */
#define WS_WEFF_OFF  (P_*KP_*2)               /* float weff[25], beff        */
#define WS_FACT_OFF  (WS_WEFF_OFF + 128)      /* float [8][400]              */

/* ---- prep: fc2 weights -> bf16, K-padded, row-major [q][k] (B^T layout) -- */
__global__ void k_prep_w(const float* __restrict__ fc2w, bf16* __restrict__ wb) {
  int idx = blockIdx.x * blockDim.x + threadIdx.x;
  if (idx >= P_ * KP_) return;
  int q = idx / KP_, k = idx % KP_;
  float v = (k < P_) ? fc2w[q * P_ + k] : 0.f;
  wb[idx] = (bf16)v;
}

/* ---- prep: fold conv1 -> conv2 -> channel-mean into one 5x5 kernel ------ */
__global__ void k_prep_conv(const float* __restrict__ c1w, const float* __restrict__ c1b,
                            const float* __restrict__ c2w, const float* __restrict__ c2b,
                            float* __restrict__ weff) {
  if (threadIdx.x != 0 || blockIdx.x != 0) return;
  float w2m[5][3][3];
  float b2m = 0.f;
  for (int o2 = 0; o2 < 10; o2++) b2m += c2b[o2];
  b2m *= 0.1f;
  for (int o1 = 0; o1 < 5; o1++)
    for (int i = 0; i < 3; i++)
      for (int j = 0; j < 3; j++) {
        float s = 0.f;
        for (int o2 = 0; o2 < 10; o2++) s += c2w[((o2 * 5 + o1) * 3 + i) * 3 + j];
        w2m[o1][i][j] = s * 0.1f;
      }
  float we[5][5];
  for (int a = 0; a < 5; a++) for (int b = 0; b < 5; b++) we[a][b] = 0.f;
  float beff = b2m;
  for (int o1 = 0; o1 < 5; o1++) {
    float sw2 = 0.f;
    for (int i2 = 0; i2 < 3; i2++)
      for (int j2 = 0; j2 < 3; j2++) {
        float wv = w2m[o1][i2][j2];
        sw2 += wv;
        for (int i1 = 0; i1 < 3; i1++)
          for (int j1 = 0; j1 < 3; j1++)
            we[i1 + i2][j1 + j2] += c1w[o1 * 9 + i1 * 3 + j1] * wv;
      }
    beff += c1b[o1] * sw2;
  }
  for (int a = 0; a < 5; a++) for (int b = 0; b < 5; b++) weff[a * 5 + b] = we[a][b];
  weff[25] = beff;
}

/* ---- main fused kernel: fc1+LN1 -> WMMA fc2 -> LN2 -> 5x5 conv -> sigmoid,
        transposed (B,P,S) coalesced store ---------------------------------- */
__global__ void __launch_bounds__(256)
k_main(const float* __restrict__ x0,
       const float* __restrict__ fc1w, const float* __restrict__ fc1b,
       const float* __restrict__ ln1g, const float* __restrict__ ln1b,
       const bf16*  __restrict__ wB,   const float* __restrict__ fc2b,
       const float* __restrict__ ln2g, const float* __restrict__ ln2b,
       const float* __restrict__ weff, float* __restrict__ out)
{
  __shared__ __align__(16) bf16 h1[TS_ * KP_];   /* 26.6 KB, WMMA A tile   */
  __shared__ __align__(16) bf16 h2[TS_ * KP_];   /* 26.6 KB, GEMM out tile */

  const int tid   = threadIdx.x;
  const int sbase = blockIdx.x * TS_;            /* flat (b,s) sample base */

  /* 8 threads per sample; LN stats by 8-lane shuffle butterfly (no atomics) */
  const int sl = tid >> 3;          /* 0..31 local sample  */
  const int pl = tid & 7;           /* 0..7 p-phase        */
  {
    const float* xp = x0 + (size_t)(sbase + sl) * D_;
    const float xa = xp[0], xb = xp[1], xc = xp[2];
    float s1 = 0.f, s2 = 0.f;
#pragma unroll 5
    for (int k = 0; k < 50; k++) {
      int p = pl + 8 * k;
      float r = fc1w[p * 3 + 0] * xa + fc1w[p * 3 + 1] * xb +
                fc1w[p * 3 + 2] * xc + fc1b[p];
      r = fmaxf(r, 0.f);
      s1 += r; s2 += r * r;
    }
#pragma unroll
    for (int off = 1; off < 8; off <<= 1) {
      s1 += __shfl_xor(s1, off, 32);
      s2 += __shfl_xor(s2, off, 32);
    }
    float mu = s1 * (1.f / P_);
    float rs = rsqrtf(s2 * (1.f / P_) - mu * mu + 1e-5f);
#pragma unroll 5
    for (int k = 0; k < 50; k++) {
      int p = pl + 8 * k;
      float r = fc1w[p * 3 + 0] * xa + fc1w[p * 3 + 1] * xb +
                fc1w[p * 3 + 2] * xc + fc1b[p];
      r = fmaxf(r, 0.f);
      h1[sl * KP_ + p] = (bf16)((r - mu) * rs * ln1g[p] + ln1b[p]);
    }
    if (pl < 2) {                   /* zero the 16-column K pad */
#pragma unroll
      for (int k = 0; k < 8; k++) h1[sl * KP_ + P_ + pl * 8 + k] = (bf16)0.f;
    }
  }
  __syncthreads();

  /* fc2 GEMM via v_wmma_f32_16x16x32_bf16.
     Jobs over 25 N-tiles; each job computes BOTH 16-row M-tiles so every
     B fragment is loaded once and feeds two WMMAs (26 wmma / job). */
  const int wave = tid >> 5, lane = tid & 31;
  const int ln = lane & 15, lh = lane >> 4;
  for (int nt = wave; nt < P_ / 16; nt += 8) {
    int q = nt * 16 + ln;
    v8f acc0 = {}, acc1 = {};
    for (int kk = 0; kk < KP_ / 32; kk++) {
      int kb = kk * 32 + lh * 8;
      const v8bf* bl  = (const v8bf*)&wB[(size_t)q * KP_ + kb];
      const v8bf* bh  = (const v8bf*)&wB[(size_t)q * KP_ + kb + 16];
      const v8bf* a0l = (const v8bf*)&h1[ln * KP_ + kb];
      const v8bf* a0h = (const v8bf*)&h1[ln * KP_ + kb + 16];
      const v8bf* a1l = (const v8bf*)&h1[(16 + ln) * KP_ + kb];
      const v8bf* a1h = (const v8bf*)&h1[(16 + ln) * KP_ + kb + 16];
      v16bf a0, a1, b;
#pragma unroll
      for (int i = 0; i < 8; i++) {
        b[i]  = (*bl)[i];  b[i + 8]  = (*bh)[i];
        a0[i] = (*a0l)[i]; a0[i + 8] = (*a0h)[i];
        a1[i] = (*a1l)[i]; a1[i + 8] = (*a1h)[i];
      }
      acc0 = __builtin_amdgcn_wmma_f32_16x16x32_bf16(false, a0, false, b,
                                                     (short)0, acc0, false, false);
      acc1 = __builtin_amdgcn_wmma_f32_16x16x32_bf16(false, a1, false, b,
                                                     (short)0, acc1, false, false);
    }
    float bias = fc2b[q];
#pragma unroll
    for (int r = 0; r < 8; r++) {               /* C layout: M = lh*8 + r, N = ln */
      int m = lh * 8 + r;
      h2[m * KP_ + q]        = (bf16)fmaxf(acc0[r] + bias, 0.f);
      h2[(16 + m) * KP_ + q] = (bf16)fmaxf(acc1[r] + bias, 0.f);
    }
  }
  __syncthreads();

  /* LN2: same 8-lane butterfly scheme on the bf16 tile */
  {
    float s1 = 0.f, s2 = 0.f;
    float vv[50];
#pragma unroll 5
    for (int k = 0; k < 50; k++) {
      float v = (float)h2[sl * KP_ + pl + 8 * k];
      vv[k] = v; s1 += v; s2 += v * v;
    }
#pragma unroll
    for (int off = 1; off < 8; off <<= 1) {
      s1 += __shfl_xor(s1, off, 32);
      s2 += __shfl_xor(s2, off, 32);
    }
    float mu = s1 * (1.f / P_);
    float rs = rsqrtf(s2 * (1.f / P_) - mu * mu + 1e-5f);
#pragma unroll 5
    for (int k = 0; k < 50; k++) {
      int p = pl + 8 * k;
      h2[sl * KP_ + p] = (bf16)((vv[k] - mu) * rs * ln2g[p] + ln2b[p]);
    }
  }
  __syncthreads();

  /* folded 5x5 circular conv + sigmoid; transposed store: lanes = consecutive s */
  float wk[26];
#pragma unroll
  for (int i = 0; i < 26; i++) wk[i] = weff[i];
  for (int idx = tid; idx < TS_ * P_; idx += 256) {
    int p  = idx >> 5;                 /* TS_ == 32 */
    int sp = idx & 31;
    int y = p / 20, x = p % 20;
    float v = wk[25];
#pragma unroll
    for (int a = 0; a < 5; a++) {
      int yy = y + a - 2; yy += (yy < 0) ? 20 : 0; yy -= (yy >= 20) ? 20 : 0;
#pragma unroll
      for (int b = 0; b < 5; b++) {
        int xx = x + b - 2; xx += (xx < 0) ? 20 : 0; xx -= (xx >= 20) ? 20 : 0;
        v += wk[a * 5 + b] * (float)h2[sp * KP_ + yy * 20 + xx];
      }
    }
    v = 1.f / (1.f + __expf(-v));
    int sg = sbase + sp;
    int bb = sg >> 12, s = sg & (S_ - 1);
    out[((size_t)bb * P_ + p) * S_ + s] = v;
  }
}

/* ---- greedy residual-suppression selection, one block per batch --------- */
__global__ void __launch_bounds__(1024)
k_select(const float* __restrict__ x0, const float* __restrict__ w,
         float* __restrict__ factor)
{
  __shared__ __align__(16) float mS[S_];    /* 16 KB: mean_d(x0)        */
  __shared__ __align__(16) float res[S_];   /* 16 KB: residual           */
  __shared__ __align__(16) float rm2[S_];   /* 16 KB: (res*m)^2          */
  __shared__ float aliveL[P_];
  __shared__ float factL[P_];
  __shared__ float actL[P_];
  __shared__ int   sPid;
  __shared__ float sScale, sMean;

  const int b   = blockIdx.x;
  const int tid = threadIdx.x;
  const float* wb = w + (size_t)b * P_ * S_;

  for (int s = tid; s < S_; s += 1024) {
    const float* xp = x0 + ((size_t)b * S_ + s) * 3;
    mS[s]  = (xp[0] + xp[1] + xp[2]) * (1.f / 3.f);
    res[s] = 1.f;
  }
  for (int p = tid; p < P_; p += 1024) { aliveL[p] = 1.f; factL[p] = INHIB_; }
  __syncthreads();

  const int wave = tid >> 5, lane = tid & 31;
  for (int it = 0; it < P_; it++) {
    /* q[s] = (res*m)^2 once per iteration (exactly 1 float4 per thread) */
    {
      v4f rv = ((const v4f*)res)[tid];
      v4f mv = ((const v4f*)mS)[tid];
      v4f qv;
#pragma unroll
      for (int c = 0; c < 4; c++) { float t = rv[c] * mv[c]; qv[c] = t * t; }
      ((v4f*)rm2)[tid] = qv;
    }
    __syncthreads();

    /* act[p] = sum_s w[p,s]^2 * q[s] ; b128 row scan + prefetch of next row */
    for (int p = wave; p < P_; p += 32) {
      const v4f* wr4 = (const v4f*)(wb + (size_t)p * S_);
      const char* wnext = (const char*)(wb + (size_t)((p + 32 < P_) ? p + 32 : p) * S_);
#pragma unroll
      for (int j = 0; j < 4; j++)                 /* 32 lanes x 4 x 128B = 16KB */
        __builtin_prefetch(wnext + ((size_t)lane + 32 * j) * 128, 0, 0);
      float s2 = 0.f;
      for (int i = lane; i < S_ / 4; i += 32) {
        v4f wv = wr4[i];
        v4f qv = ((const v4f*)rm2)[i];
#pragma unroll
        for (int c = 0; c < 4; c++) s2 += wv[c] * wv[c] * qv[c];
      }
#pragma unroll
      for (int off = 16; off > 0; off >>= 1) s2 += __shfl_xor(s2, off, 32);
      if (lane == 0) actL[p] = aliveL[p] * s2;
    }
    __syncthreads();

    /* argmax with first-max tie rule (matches jnp.argmax) */
    if (wave == 0) {
      float bv = -1.f; int bi = 0x7fffffff;
      for (int p = lane; p < P_; p += 32) {
        float v = actL[p];
        if (v > bv) { bv = v; bi = p; }
      }
#pragma unroll
      for (int off = 16; off > 0; off >>= 1) {
        float ov = __shfl_xor(bv, off, 32);
        int   oi = __shfl_xor(bi, off, 32);
        if (ov > bv || (ov == bv && oi < bi)) { bv = ov; bi = oi; }
      }
      if (lane == 0) {
        sPid   = bi;
        sScale = aliveL[bi];
        aliveL[bi] = 0.f;
        factL[bi]  = 1.f;
        sMean  = 0.f;
      }
    }
    __syncthreads();

    /* residual update: exactly one float4 per thread */
    const int pid = sPid;
    const float scale = sScale;
    const v4f* wr4 = (const v4f*)(wb + (size_t)pid * S_);
    float part = 0.f;
    {
      v4f rv = ((const v4f*)res)[tid];
      v4f wv = wr4[tid];
#pragma unroll
      for (int c = 0; c < 4; c++) {
        float nr = fmaxf(rv[c] - scale * wv[c], 0.f);
        rv[c] = nr; part += nr;
      }
      ((v4f*)res)[tid] = rv;
    }
#pragma unroll
    for (int off = 16; off > 0; off >>= 1) part += __shfl_xor(part, off, 32);
    if (lane == 0) atomicAdd(&sMean, part);
    __syncthreads();
    if (sMean * (1.f / S_) < ENDRATE_) break;   /* uniform: all threads agree */
  }
  __syncthreads();
  for (int p = tid; p < P_; p += 1024) factor[b * P_ + p] = factL[p];
}

/* ---- out *= bursting (float4; factor invariant within a float4) --------- */
__global__ void k_scale(float* __restrict__ out, const float* __restrict__ factor) {
  size_t i4 = (size_t)blockIdx.x * blockDim.x + threadIdx.x;
  if (i4 >= (size_t)B_ * P_ * S_ / 4) return;
  float f = factor[i4 >> 10];       /* (i4*4)/S_ == b*P_+p */
  v4f v = ((const v4f*)out)[i4];
#pragma unroll
  for (int c = 0; c < 4; c++) v[c] *= f;
  ((v4f*)out)[i4] = v;
}

extern "C" void kernel_launch(void* const* d_in, const int* in_sizes, int n_in,
                              void* d_out, int out_size, void* d_ws, size_t ws_size,
                              hipStream_t stream)
{
  const float* x0   = (const float*)d_in[0];
  const float* fc1w = (const float*)d_in[1];
  const float* fc1b = (const float*)d_in[2];
  const float* ln1g = (const float*)d_in[3];
  const float* ln1b = (const float*)d_in[4];
  const float* fc2w = (const float*)d_in[5];
  const float* fc2b = (const float*)d_in[6];
  const float* ln2g = (const float*)d_in[7];
  const float* ln2b = (const float*)d_in[8];
  const float* c1w  = (const float*)d_in[9];
  const float* c1b  = (const float*)d_in[10];
  const float* c2w  = (const float*)d_in[11];
  const float* c2b  = (const float*)d_in[12];
  float* out = (float*)d_out;

  bf16*  wB   = (bf16*)d_ws;
  float* weff = (float*)((char*)d_ws + WS_WEFF_OFF);
  float* fact = (float*)((char*)d_ws + WS_FACT_OFF);

  k_prep_w   <<<(P_ * KP_ + 255) / 256, 256, 0, stream>>>(fc2w, wB);
  k_prep_conv<<<1, 32, 0, stream>>>(c1w, c1b, c2w, c2b, weff);
  k_main     <<<(B_ * S_) / TS_, 256, 0, stream>>>(x0, fc1w, fc1b, ln1g, ln1b,
                                                   wB, fc2b, ln2g, ln2b, weff, out);
  k_select   <<<B_, 1024, 0, stream>>>(x0, out, fact);
  k_scale    <<<((size_t)B_ * P_ * S_ / 4 + 255) / 256, 256, 0, stream>>>(out, fact);
}
